// TransformerBlock_29858612641998
// MI455X (gfx1250) — matrix operand order repaired and checked by reading, not compiled, not run on hardware
//
#include <hip/hip_runtime.h>
#include <math.h>
#include <stdint.h>

// ---------------- CDNA5 WMMA types ----------------
typedef __bf16 bf16;
typedef __attribute__((ext_vector_type(16))) __bf16 v16bf;
typedef __attribute__((ext_vector_type(8)))  __bf16 v8bf;
typedef __attribute__((ext_vector_type(8)))  float  v8f;

#define DIM_   1024
#define NH_    16
#define HD_    64
#define FF_    4096
#define B_     2
#define S_     2048
#define MROWS_ (B_ * S_)   // 4096

__device__ __forceinline__ v8f v8f_zero() {
    v8f z;
#pragma unroll
    for (int i = 0; i < 8; ++i) z[i] = 0.0f;
    return z;
}

__device__ __forceinline__ v16bf cat8(v8bf lo, v8bf hi) {
    v16bf r;
#pragma unroll
    for (int e = 0; e < 8; ++e) { r[e] = lo[e]; r[e + 8] = hi[e]; }
    return r;
}

// ---- CDNA5 async global->LDS (ASYNCcnt-tracked, no VGPR payload) ----
// lds_addr: wave-relative LDS byte offset (low 32 bits of a generic LDS pointer).
__device__ __forceinline__ void async_load_b128(uint32_t lds_addr, const void* gaddr) {
    asm volatile("global_load_async_to_lds_b128 %0, %1, off"
                 :: "v"(lds_addr), "v"(gaddr) : "memory");
}
#define WAIT_ASYNC(n) asm volatile("s_wait_asynccnt " #n ::: "memory")

__device__ __forceinline__ uint32_t lds_off(const void* p) {
    return (uint32_t)(uintptr_t)p;   // LDS aperture: addr[31:0] is the LDS offset
}

// ---------------- fp32 -> bf16 conversion ----------------
__global__ void cvt_f32_bf16(const float* __restrict__ in, bf16* __restrict__ out, int n) {
    int i = blockIdx.x * blockDim.x + threadIdx.x;
    if (i < n) out[i] = (bf16)in[i];
}

// ---------------- Generic bf16 WMMA GEMM: C[M,N] = A[M,K] * W[N,K]^T ----------------
// Block: 256 threads (8 waves). Block tile 256x128, BK=32, double-buffered async staging.
// Wave tile 64x64 -> 4x4 WMMA tiles (16 accumulators), 16 ds_load_b128 per 16 WMMAs.
__global__ __launch_bounds__(256) void gemm_bf16_wmma(
    const bf16* __restrict__ A, const bf16* __restrict__ W,
    float* __restrict__ C, int M, int N, int K)
{
    __shared__ __attribute__((aligned(16))) bf16 As[2][256 * 32];
    __shared__ __attribute__((aligned(16))) bf16 Ws[2][128 * 32];

    const int tid  = threadIdx.x;
    const int m0   = blockIdx.y * 256;
    const int n0   = blockIdx.x * 128;
    const int wave = tid >> 5;
    const int lane = tid & 31;
    const int wm   = (wave >> 1) * 64;   // 0,64,128,192
    const int wn   = (wave & 1) * 64;    // 0,64
    const int half = lane >> 4;          // 0/1
    const int l16  = lane & 15;

    const uint32_t asb = lds_off(&As[0][0]);
    const uint32_t wsb = lds_off(&Ws[0][0]);

    v8f acc[4][4];
#pragma unroll
    for (int a = 0; a < 4; ++a)
#pragma unroll
        for (int b = 0; b < 4; ++b) acc[a][b] = v8f_zero();

    const int arow = tid;              // A stage: 1 row (32 elems / 64 B) per thread
    const int brow = tid >> 1;         // W stage: half row per thread
    const int bseg = (tid & 1) * 16;

    const int nk = K / 32;

    // per wave per tile: 6 async instructions
    auto issue = [&](int buf, int k0) {
        const bf16* ag = &A[(size_t)(m0 + arow) * K + k0];
        const uint32_t al = asb + (uint32_t)(buf * (256 * 32) + arow * 32) * 2;
        async_load_b128(al,      ag);
        async_load_b128(al + 16, ag + 8);
        async_load_b128(al + 32, ag + 16);
        async_load_b128(al + 48, ag + 24);
        const bf16* wg = &W[(size_t)(n0 + brow) * K + k0 + bseg];
        const uint32_t wl = wsb + (uint32_t)(buf * (128 * 32) + brow * 32 + bseg) * 2;
        async_load_b128(wl,      wg);
        async_load_b128(wl + 16, wg + 8);
    };

    issue(0, 0);
    for (int kt = 0; kt < nk; ++kt) {
        const int buf = kt & 1;
        if (kt + 1 < nk) {
            issue(buf ^ 1, (kt + 1) * 32);  // prefetch next tile into other buffer
            WAIT_ASYNC(6);                  // oldest 6 (current tile) complete, in order
        } else {
            WAIT_ASYNC(0);
        }
        __syncthreads();

        // A fragments: 16x32, lane(m=l16, half): elems 0..7 -> K=8*half..+7,
        // elems 8..15 -> K=16+8*half..+7.
        v16bf af[4];
#pragma unroll
        for (int a = 0; a < 4; ++a) {
            const int r = wm + a * 16 + l16;
            v8bf lo = *(const v8bf*)&As[buf][r * 32 + 8 * half];
            v8bf hi = *(const v8bf*)&As[buf][r * 32 + 16 + 8 * half];
            af[a] = cat8(lo, hi);
        }
#pragma unroll
        for (int bb = 0; bb < 4; ++bb) {
            // B fragment: 32x16, lane(n=l16, half): elems 0..7 -> K=8*half..+7, elems 8..15 -> K=16+8*half..+7.
            const int n = wn + bb * 16 + l16;
            v8bf lo = *(const v8bf*)&Ws[buf][n * 32 + 8 * half];
            v8bf hi = *(const v8bf*)&Ws[buf][n * 32 + 16 + 8 * half];
            v16bf bfv = cat8(lo, hi);
#pragma unroll
            for (int a = 0; a < 4; ++a)
                acc[a][bb] = __builtin_amdgcn_wmma_f32_16x16x32_bf16(
                    false, af[a], false, bfv, (short)0, acc[a][bb], false, false);
        }
        __syncthreads();
    }

    // Epilogue: C layout -> VGPR r holds row (8*half + r), col = l16 within tile.
#pragma unroll
    for (int a = 0; a < 4; ++a)
#pragma unroll
        for (int bb = 0; bb < 4; ++bb)
#pragma unroll
            for (int r = 0; r < 8; ++r) {
                const int row = m0 + wm + a * 16 + 8 * half + r;
                const int col = n0 + wn + bb * 16 + l16;
                C[(size_t)row * N + col] = acc[a][bb][r];
            }
}

// ---------------- RoPE + QKV split to [B,H,S,64] bf16 ----------------
__global__ void rope_split(const float* __restrict__ qkv,
                           bf16* __restrict__ q, bf16* __restrict__ k, bf16* __restrict__ v)
{
    const int idx = blockIdx.x * blockDim.x + threadIdx.x;  // 2^21 threads
    const int i  = idx & 31;
    const int h  = (idx >> 5) & 15;
    const int s  = (idx >> 9) & (S_ - 1);
    const int b  = idx >> 20;

    const size_t row = (size_t)(b * S_ + s) * (3 * DIM_);
    const float* qp = qkv + row + h * HD_;
    const float* kp = qkv + row + DIM_ + h * HD_;
    const float* vp = qkv + row + 2 * DIM_ + h * HD_;

    const float inv = __powf(10000.0f, -(float)(2 * i) / (float)HD_);
    const float ang = (float)s * inv;
    const float c = __cosf(ang), sn = __sinf(ang);

    const float q1 = qp[i], q2 = qp[i + 32];
    const float k1 = kp[i], k2 = kp[i + 32];

    const size_t o = ((size_t)(b * NH_ + h) * S_ + s) * HD_;
    q[o + i]      = (bf16)(q1 * c - q2 * sn);
    q[o + i + 32] = (bf16)(q1 * sn + q2 * c);
    k[o + i]      = (bf16)(k1 * c - k2 * sn);
    k[o + i + 32] = (bf16)(k1 * sn + k2 * c);
    v[o + i]      = (bf16)vp[i];
    v[o + i + 32] = (bf16)vp[i + 32];
}

// ---------------- Flash-style causal attention (per wave: 16 q rows) ----------------
// grid = (S/128, B*H), block = 256 (8 waves). Key tiles of 32 staged in LDS.
__global__ __launch_bounds__(256) void attn_wmma(
    const bf16* __restrict__ Qg, const bf16* __restrict__ Kg,
    const bf16* __restrict__ Vg, bf16* __restrict__ Og)
{
    __shared__ __attribute__((aligned(16))) bf16 Ks[32 * 64];     // [key][feat]
    __shared__ __attribute__((aligned(16))) bf16 Vt[64 * 32];     // [feat][key]
    __shared__ __attribute__((aligned(16))) bf16 Ps[8 * 16 * 32]; // per-wave P patch

    const int tid  = threadIdx.x;
    const int wave = tid >> 5;
    const int lane = tid & 31;
    const int half = lane >> 4;
    const int l16  = lane & 15;

    const int qb = blockIdx.x * 128;
    const int bh = blockIdx.y;
    const bf16* Q = Qg + (size_t)bh * S_ * HD_;
    const bf16* K = Kg + (size_t)bh * S_ * HD_;
    const bf16* V = Vg + (size_t)bh * S_ * HD_;

    const uint32_t ksb = lds_off(&Ks[0]);

    // Q fragments (features 0..31 and 32..63)
    v16bf qf[2];
    {
        const int row = qb + wave * 16 + l16;
#pragma unroll
        for (int f = 0; f < 2; ++f) {
            v8bf lo = *(const v8bf*)&Q[(size_t)row * HD_ + f * 32 + 8 * half];
            v8bf hi = *(const v8bf*)&Q[(size_t)row * HD_ + f * 32 + 16 + 8 * half];
            qf[f] = cat8(lo, hi);
        }
    }

    float mrow[8], lrow[8];
    v8f o[4];
#pragma unroll
    for (int r = 0; r < 8; ++r) { mrow[r] = -3.0e38f; lrow[r] = 0.0f; }
#pragma unroll
    for (int t = 0; t < 4; ++t) o[t] = v8f_zero();

    const float scale = 0.125f;  // 1/sqrt(64)
    const int jtiles = qb / 32 + 4;

    for (int jt = 0; jt < jtiles; ++jt) {
        const int kbase = jt * 32;
        // Stage K tile [32][64] via async load; V^T tile [64][32] via VGPR transpose.
        {
            const int r = tid >> 3;            // 0..31 key row
            const int c = (tid & 7) * 8;       // feature chunk
            async_load_b128(ksb + (uint32_t)(r * 64 + c) * 2,
                            &K[(size_t)(kbase + r) * HD_ + c]);
            v8bf vv = *(const v8bf*)&V[(size_t)(kbase + r) * HD_ + c];
#pragma unroll
            for (int e = 0; e < 8; ++e) Vt[(c + e) * 32 + r] = vv[e];
        }
        WAIT_ASYNC(0);
        __syncthreads();

        // scores: Q(16x64) @ K^T(64x32) -> two 16x16 C tiles
        v8f sc[2];
#pragma unroll
        for (int ct = 0; ct < 2; ++ct) {
            v8f s = v8f_zero();
#pragma unroll
            for (int f = 0; f < 2; ++f) {
                const int kr = ct * 16 + l16;
                v8bf lo = *(const v8bf*)&Ks[kr * 64 + f * 32 + 8 * half];
                v8bf hi = *(const v8bf*)&Ks[kr * 64 + f * 32 + 16 + 8 * half];
                v16bf kb = cat8(lo, hi);
                s = __builtin_amdgcn_wmma_f32_16x16x32_bf16(
                        false, qf[f], false, kb, (short)0, s, false, false);
            }
            sc[ct] = s;
        }

        // Online softmax per row, P staged to LDS as bf16
        const int qrow_base = qb + wave * 16 + 8 * half;
#pragma unroll
        for (int r = 0; r < 8; ++r) {
            const int qi = qrow_base + r;
            float s0 = sc[0][r] * scale;
            float s1 = sc[1][r] * scale;
            if (kbase + l16 > qi)      s0 = -1.0e9f;
            if (kbase + 16 + l16 > qi) s1 = -1.0e9f;
            float t = fmaxf(s0, s1);
#pragma unroll
            for (int off = 8; off; off >>= 1) t = fmaxf(t, __shfl_xor(t, off, 16));
            const float mnew = fmaxf(mrow[r], t);
            const float p0 = __expf(s0 - mnew);
            const float p1 = __expf(s1 - mnew);
            float rs = p0 + p1;
#pragma unroll
            for (int off = 8; off; off >>= 1) rs += __shfl_xor(rs, off, 16);
            const float corr = __expf(mrow[r] - mnew);
            lrow[r] = lrow[r] * corr + rs;
            mrow[r] = mnew;
#pragma unroll
            for (int t4 = 0; t4 < 4; ++t4) o[t4][r] *= corr;
            Ps[wave * 512 + (8 * half + r) * 32 + l16]      = (bf16)p0;
            Ps[wave * 512 + (8 * half + r) * 32 + 16 + l16] = (bf16)p1;
        }

        // Reload P as A fragment (16x32)
        v16bf pf;
        {
            v8bf lo = *(const v8bf*)&Ps[wave * 512 + l16 * 32 + 8 * half];
            v8bf hi = *(const v8bf*)&Ps[wave * 512 + l16 * 32 + 16 + 8 * half];
            pf = cat8(lo, hi);
        }
        // O += P(16x32) @ V(32x64): 4 feature tiles
#pragma unroll
        for (int t4 = 0; t4 < 4; ++t4) {
            const int feat = t4 * 16 + l16;
            v8bf lo = *(const v8bf*)&Vt[feat * 32 + 8 * half];
            v8bf hi = *(const v8bf*)&Vt[feat * 32 + 16 + 8 * half];
            v16bf vb = cat8(lo, hi);
            o[t4] = __builtin_amdgcn_wmma_f32_16x16x32_bf16(
                        false, pf, false, vb, (short)0, o[t4], false, false);
        }
        __syncthreads();
    }

    // Finalize: write [B,S,H*64] bf16 for the attn_out GEMM
    const int b = bh >> 4, h = bh & 15;
#pragma unroll
    for (int t4 = 0; t4 < 4; ++t4)
#pragma unroll
        for (int r = 0; r < 8; ++r) {
            const int qi = qb + wave * 16 + 8 * half + r;
            const float val = o[t4][r] / lrow[r];
            Og[(size_t)(b * S_ + qi) * DIM_ + h * HD_ + t4 * 16 + l16] = (bf16)val;
        }
}

// ---------------- residual add + RMS norm (one row per block) ----------------
__global__ __launch_bounds__(256) void add_rmsnorm(
    const float* __restrict__ a, const float* __restrict__ b,
    const float* __restrict__ scale, float* __restrict__ outf, bf16* __restrict__ outb)
{
    __shared__ float red[8];
    __shared__ float stot;
    const int row = blockIdx.x;
    const int tid = threadIdx.x;

    float x[4];
    float ss = 0.0f;
#pragma unroll
    for (int i = 0; i < 4; ++i) {
        const int c = tid + i * 256;
        x[i] = a[(size_t)row * DIM_ + c] + b[(size_t)row * DIM_ + c];
        ss += x[i] * x[i];
    }
#pragma unroll
    for (int off = 16; off; off >>= 1) ss += __shfl_xor(ss, off, 32);
    if ((tid & 31) == 0) red[tid >> 5] = ss;
    __syncthreads();
    if (tid == 0) {
        float t = 0.0f;
        for (int i = 0; i < 8; ++i) t += red[i];
        stot = t;
    }
    __syncthreads();
    const float norm = sqrtf(stot) * 0.03125f;  // * 1024^-0.5
    const float inv = 1.0f / (norm + 1e-8f);
#pragma unroll
    for (int i = 0; i < 4; ++i) {
        const int c = tid + i * 256;
        const float y = scale[c] * x[i] * inv;
        outf[(size_t)row * DIM_ + c] = y;
        if (outb) outb[(size_t)row * DIM_ + c] = (bf16)y;
    }
}

// ---------------- sigmoid(gate+gb) * (lin+lb) -> bf16 ----------------
__global__ void gated_act(const float* __restrict__ g, const float* __restrict__ l,
                          const float* __restrict__ gb, const float* __restrict__ lb,
                          bf16* __restrict__ out, int n, int cols)
{
    const int i = blockIdx.x * blockDim.x + threadIdx.x;
    if (i < n) {
        const int c = i & (cols - 1);
        const float gg = g[i] + gb[c];
        const float ll = l[i] + lb[c];
        out[i] = (bf16)(ll / (1.0f + __expf(-gg)));
    }
}

// ---------------- launch ----------------
extern "C" void kernel_launch(void* const* d_in, const int* in_sizes, int n_in,
                              void* d_out, int out_size, void* d_ws, size_t ws_size,
                              hipStream_t stream) {
    const float* x_in   = (const float*)d_in[0];
    // d_in[1] = mask (causal applied analytically)
    const float* qkv_w  = (const float*)d_in[2];
    const float* ao_w   = (const float*)d_in[3];
    const float* gate_w = (const float*)d_in[4];
    const float* gate_b = (const float*)d_in[5];
    const float* lin_w  = (const float*)d_in[6];
    const float* lin_b  = (const float*)d_in[7];
    const float* ffo_w  = (const float*)d_in[8];
    const float* n1_s   = (const float*)d_in[9];
    const float* n2_s   = (const float*)d_in[10];

    char* base = (char*)d_ws;
    const size_t MB = 1ull << 20;
    bf16*  w_qkv = (bf16*)(base + 0);         //  6 MB [3072,1024]
    bf16*  w_ao  = (bf16*)(base + 6  * MB);   //  2 MB [1024,1024]
    bf16*  w_gat = (bf16*)(base + 8  * MB);   //  8 MB [4096,1024]
    bf16*  w_lin = (bf16*)(base + 16 * MB);   //  8 MB [4096,1024]
    bf16*  w_ffo = (bf16*)(base + 24 * MB);   //  8 MB [1024,4096]
    bf16*  xb    = (bf16*)(base + 32 * MB);   //  8 MB [4096,1024]
    float* h1f   = (float*)(base + 40 * MB);  // 16 MB [4096,1024]
    bf16*  h1b   = (bf16*)(base + 56 * MB);   //  8 MB
    char*  big   = base + 64 * MB;
    float* qkvf  = (float*)big;               // 48 MB [4096,3072]
    bf16*  q16   = (bf16*)(big + 48 * MB);    //  8 MB [B,H,S,64]
    bf16*  k16   = (bf16*)(big + 56 * MB);    //  8 MB
    bf16*  v16   = (bf16*)(big + 64 * MB);    //  8 MB
    bf16*  ao_in = (bf16*)(big + 72 * MB);    //  8 MB [4096,1024]
    float* projf = (float*)big;               // 16 MB (reuse: qkvf dead)
    float* gatef = (float*)big;               // 64 MB (reuse: projf dead)
    float* linf  = (float*)(big + 64 * MB);   // 64 MB (reuse: q/k/v/ao_in dead)
    bf16*  ff16  = (bf16*)(big + 128 * MB);   // 32 MB [4096,4096]
    float* fff   = (float*)big;               // 16 MB (reuse: gatef dead)

    const dim3 blk(256);
    auto cvt = [&](const float* src, bf16* dst, int n) {
        cvt_f32_bf16<<<(n + 255) / 256, blk, 0, stream>>>(src, dst, n);
    };
    cvt(x_in,   xb,    MROWS_ * DIM_);
    cvt(qkv_w,  w_qkv, 3 * DIM_ * DIM_);
    cvt(ao_w,   w_ao,  DIM_ * DIM_);
    cvt(gate_w, w_gat, FF_ * DIM_);
    cvt(lin_w,  w_lin, FF_ * DIM_);
    cvt(ffo_w,  w_ffo, DIM_ * FF_);

    // 1) QKV projection: [4096,1024] x [3072,1024]^T
    gemm_bf16_wmma<<<dim3(3 * DIM_ / 128, MROWS_ / 256), blk, 0, stream>>>(
        xb, w_qkv, qkvf, MROWS_, 3 * DIM_, DIM_);

    // 2) RoPE + split
    rope_split<<<(B_ * S_ * NH_ * 32) / 256, blk, 0, stream>>>(qkvf, q16, k16, v16);

    // 3) Causal attention
    attn_wmma<<<dim3(S_ / 128, B_ * NH_), blk, 0, stream>>>(q16, k16, v16, ao_in);

    // 4) Output projection
    gemm_bf16_wmma<<<dim3(DIM_ / 128, MROWS_ / 256), blk, 0, stream>>>(
        ao_in, w_ao, projf, MROWS_, DIM_, DIM_);

    // 5) residual + RMSNorm 1
    add_rmsnorm<<<MROWS_, blk, 0, stream>>>(x_in, projf, n1_s, h1f, h1b);

    // 6) FF gate / lin GEMMs
    gemm_bf16_wmma<<<dim3(FF_ / 128, MROWS_ / 256), blk, 0, stream>>>(
        h1b, w_gat, gatef, MROWS_, FF_, DIM_);
    gemm_bf16_wmma<<<dim3(FF_ / 128, MROWS_ / 256), blk, 0, stream>>>(
        h1b, w_lin, linf, MROWS_, FF_, DIM_);

    // 7) sigmoid(gate) * lin
    gated_act<<<(MROWS_ * FF_) / 256, blk, 0, stream>>>(
        gatef, linf, gate_b, lin_b, ff16, MROWS_ * FF_, FF_);

    // 8) FF out projection: [4096,4096] x [1024,4096]^T
    gemm_bf16_wmma<<<dim3(DIM_ / 128, MROWS_ / 256), blk, 0, stream>>>(
        ff16, w_ffo, fff, MROWS_, DIM_, FF_);

    // 9) residual + RMSNorm 2 -> d_out (fp32)
    add_rmsnorm<<<MROWS_, blk, 0, stream>>>(h1f, fff, n2_s, (float*)d_out, (bf16*)nullptr);
}
